// MultiQueryAttention_73555609911783
// MI455X (gfx1250) — compile-verified, matrix operand order checked by reading, not run on hardware
//
#include <hip/hip_runtime.h>
#include <hip/hip_bf16.h>

#define BB 2
#define TT 2048
#define DM 1024
#define NH 16
#define DH 64
#define MM (BB * TT) // 4096

typedef __attribute__((ext_vector_type(16))) __bf16 v16bf;
typedef __attribute__((ext_vector_type(8)))  __bf16 v8bf;
typedef __attribute__((ext_vector_type(4)))  __bf16 v4bf;
typedef __attribute__((ext_vector_type(8)))  float  v8f;
typedef __attribute__((ext_vector_type(4)))  float  v4f;

__device__ __forceinline__ __bf16 f2bf(float f) {
  unsigned u = __float_as_uint(f);
  u += 0x7FFFu + ((u >> 16) & 1u);            // round-to-nearest-even
  unsigned short s = (unsigned short)(u >> 16);
  __bf16 r;
  __builtin_memcpy(&r, &s, sizeof(r));
  return r;
}

__device__ __forceinline__ v4bf pk4(v4f v) {
  v4bf r;
#pragma unroll
  for (int q = 0; q < 4; ++q) r[q] = f2bf(v[q]);
  return r;
}

// Fragment from two 16B bf16 runs (LDS or global): elems 0..7 = p0, 8..15 = p1.
__device__ __forceinline__ v16bf ld_frag(const __bf16* p0, const __bf16* p1) {
  union { v16bf v; v8bf h[2]; } u;
  u.h[0] = *(const v8bf*)p0;   // b128
  u.h[1] = *(const v8bf*)p1;
  return u.v;
}

__device__ __forceinline__ v8f wmma_bf16(v16bf a, v16bf b, v8f c) {
  return __builtin_amdgcn_wmma_f32_16x16x32_bf16(
      false, a, false, b, (short)0, c, false, false);
}

// ---------------------------------------------------------------------------
// One-time f32 -> bf16 (16 elems/thread, exact launch, n % 4096 == 0)
// ---------------------------------------------------------------------------
__global__ __launch_bounds__(256)
void cvt_f32_bf16(const float* __restrict__ in, __bf16* __restrict__ out) {
  size_t base = ((size_t)blockIdx.x * 256 + threadIdx.x) * 16;
  v4f a0 = *(const v4f*)(in + base);
  v4f a1 = *(const v4f*)(in + base + 4);
  v4f a2 = *(const v4f*)(in + base + 8);
  v4f a3 = *(const v4f*)(in + base + 12);
  union { v8bf v; v4bf h[2]; } lo, hi;
  lo.h[0] = pk4(a0); lo.h[1] = pk4(a1);
  hi.h[0] = pk4(a2); hi.h[1] = pk4(a3);
  *(v8bf*)(out + base)     = lo.v;
  *(v8bf*)(out + base + 8) = hi.v;
}

// ---------------------------------------------------------------------------
// One-time transpose + convert: in[K,N] f32 -> out[N,K] bf16. 32x32 LDS tiles.
// grid (N/32, K/32), 256 threads (32x8).
// ---------------------------------------------------------------------------
__global__ __launch_bounds__(256)
void cvt_transpose_bf16(const float* __restrict__ in, __bf16* __restrict__ out,
                        int Kd, int Nd) {
  __shared__ float t[32][33];
  const int n0 = blockIdx.x * 32, k0 = blockIdx.y * 32;
  const int x = threadIdx.x & 31, y = threadIdx.x >> 5;
#pragma unroll
  for (int j = 0; j < 4; ++j)
    t[y + j * 8][x] = in[(size_t)(k0 + y + j * 8) * Nd + (n0 + x)];
  __syncthreads();
#pragma unroll
  for (int j = 0; j < 4; ++j)
    out[(size_t)(n0 + y + j * 8) * Kd + (k0 + x)] = f2bf(t[x][y + j * 8]);
}

// ---------------------------------------------------------------------------
// GEMM: C[M,N] = A[M,K] * Wt[N,K]^T + bias[N], all-bf16 operands, f32 accum.
// 256 threads = 8 waves; block tile 128 x TN; k-step 32. No bounds checks.
// Staging is pure b128 copies (A and Wt both k-contiguous).
//   TN=128: waves 4x2, wave tile 32x64  (8 WMMA/k-step)
//   TN=64 : waves 8x1, wave tile 16x64  (4 WMMA/k-step)
// ---------------------------------------------------------------------------
template <int TN, bool OUT_BF16>
__global__ __launch_bounds__(256)
void gemm_bias_wmma(const __bf16* __restrict__ A, const __bf16* __restrict__ Wt,
                    const float* __restrict__ bias, void* __restrict__ Cv,
                    int M, int N, int K, float oscale) {
  constexpr int MI = (TN == 128) ? 2 : 1;
  __shared__ __attribute__((aligned(16))) __bf16 sA[128 * 40]; // [row][k] stride 40
  __shared__ __attribute__((aligned(16))) __bf16 sB[TN * 40];  // [col][k] stride 40

  const int tid  = threadIdx.x;
  const int lane = tid & 31;
  const int wave = tid >> 5;
  const int halo = lane >> 4;
  const int lcol = lane & 15;
  const int wm   = (TN == 128) ? (wave & 3) : wave;
  const int wn   = (TN == 128) ? (wave >> 2) : 0;
  const int rowBase = blockIdx.y * 128;
  const int colBase = blockIdx.x * TN;

  v8f acc[MI][4];
#pragma unroll
  for (int mi = 0; mi < MI; ++mi)
#pragma unroll
    for (int ni = 0; ni < 4; ++ni)
#pragma unroll
      for (int e = 0; e < 8; ++e) acc[mi][ni][e] = 0.0f;

  for (int k0 = 0; k0 < K; k0 += 32) {
    // A tile 128x32 bf16: 512 x b128 copies
#pragma unroll
    for (int i = 0; i < 2; ++i) {
      int idx = tid + i * 256;
      int r = idx >> 2, c8 = idx & 3;
      *(v8bf*)&sA[r * 40 + c8 * 8] =
          *(const v8bf*)&A[(size_t)(rowBase + r) * K + k0 + c8 * 8];
    }
    // B tile TNx32 bf16 (k-contiguous rows of Wt): TN*4 x b128 copies
#pragma unroll
    for (int i = 0; i < (TN * 4) / 256; ++i) {
      int idx = tid + i * 256;
      int c = idx >> 2, c8 = idx & 3;
      *(v8bf*)&sB[c * 40 + c8 * 8] =
          *(const v8bf*)&Wt[(size_t)(colBase + c) * K + k0 + c8 * 8];
    }
    if (k0 + 32 < K)                      // gfx1250 global_prefetch_b8
      __builtin_prefetch(&A[(size_t)(rowBase + (tid >> 1)) * K + (k0 + 32)], 0, 1);
    __syncthreads();

    v16bf af[MI], bfr[4];
#pragma unroll
    for (int mi = 0; mi < MI; ++mi) {
      int r = wm * (MI * 16) + mi * 16 + lcol;
      af[mi] = ld_frag(&sA[r * 40 + halo * 8], &sA[r * 40 + halo * 8 + 16]);
    }
#pragma unroll
    for (int ni = 0; ni < 4; ++ni) {
      int cc = wn * 64 + ni * 16 + lcol;
      bfr[ni] = ld_frag(&sB[cc * 40 + halo * 8], &sB[cc * 40 + halo * 8 + 16]);
    }
#pragma unroll
    for (int mi = 0; mi < MI; ++mi)
#pragma unroll
      for (int ni = 0; ni < 4; ++ni)
        acc[mi][ni] = wmma_bf16(af[mi], bfr[ni], acc[mi][ni]);
    __syncthreads();
  }

#pragma unroll
  for (int mi = 0; mi < MI; ++mi)
#pragma unroll
    for (int ni = 0; ni < 4; ++ni)
#pragma unroll
      for (int e = 0; e < 8; ++e) {
        int grow = rowBase + wm * (MI * 16) + mi * 16 + halo * 8 + e;
        int gcol = colBase + wn * 64 + ni * 16 + lcol;
        float v = (acc[mi][ni][e] + bias[gcol]) * oscale;
        if constexpr (OUT_BF16)
          ((__bf16*)Cv)[(size_t)grow * N + gcol] = f2bf(v);
        else
          ((float*)Cv)[(size_t)grow * N + gcol] = v;
      }
}

// ---------------------------------------------------------------------------
// MQA flash attention, all-bf16 inputs (Q pre-scaled by 1/sqrt(64) upstream).
// One wave per (b, h, 16-row q tile); causal; bf16 output.
// ---------------------------------------------------------------------------
__global__ __launch_bounds__(32)
void mqa_flash_attn(const __bf16* __restrict__ Q, const __bf16* __restrict__ Kk,
                    const __bf16* __restrict__ V, __bf16* __restrict__ O) {
  __shared__ __attribute__((aligned(16))) __bf16 sP[16 * 40]; // P tile 16x32
  __shared__ __attribute__((aligned(16))) __bf16 sV[64 * 40]; // V chunk [d][s]

  const int lane = threadIdx.x;
  const int halo = lane >> 4, lcol = lane & 15;
  const int qt = blockIdx.x, h = blockIdx.y, b = blockIdx.z;

  // Q A-fragments: two b128 loads each, no conversion
  const __bf16* qp = Q + ((size_t)(b * TT + qt * 16 + lcol)) * DM + h * DH;
  v16bf aq[2];
#pragma unroll
  for (int c = 0; c < 2; ++c)
    aq[c] = ld_frag(qp + c * 32 + halo * 8, qp + c * 32 + halo * 8 + 16);

  v8f oacc[4];
  float mrun[8], lrun[8];
#pragma unroll
  for (int dt = 0; dt < 4; ++dt)
#pragma unroll
    for (int e = 0; e < 8; ++e) oacc[dt][e] = 0.0f;
#pragma unroll
  for (int e = 0; e < 8; ++e) { mrun[e] = -3.0e38f; lrun[e] = 0.0f; }

  const int nch = ((qt * 16 + 15) >> 5) + 1;   // 32-wide key chunks (causal)
  for (int ch = 0; ch < nch; ++ch) {
    const int s0 = ch * 32;

    // ---- S = Q K^T for 16x32 chunk (two 16-col halves) ----
    v8f sc[2];
#pragma unroll
    for (int hh = 0; hh < 2; ++hh) {
      const __bf16* kp = Kk + ((size_t)(b * TT + s0 + hh * 16 + lcol)) * DH;
      v8f z;
#pragma unroll
      for (int e = 0; e < 8; ++e) z[e] = 0.0f;
      z = wmma_bf16(aq[0], ld_frag(kp + halo * 8, kp + halo * 8 + 16), z);
      sc[hh] = wmma_bf16(aq[1], ld_frag(kp + 32 + halo * 8, kp + 32 + halo * 8 + 16), z);
    }

    // ---- causal mask + online softmax (row = halo*8+e, 16-lane reductions) ----
#pragma unroll
    for (int e = 0; e < 8; ++e) {
      const int mrow = qt * 16 + halo * 8 + e;
#pragma unroll
      for (int hh = 0; hh < 2; ++hh) {
        int sg = s0 + hh * 16 + lcol;
        if (sg > mrow) sc[hh][e] = -1.0e10f;  // matches reference mask value
      }
      float tmax = fmaxf(sc[0][e], sc[1][e]);
#pragma unroll
      for (int off = 1; off <= 8; off <<= 1)
        tmax = fmaxf(tmax, __shfl_xor(tmax, off, 32));
      float mnew  = fmaxf(mrun[e], tmax);
      float scale = __expf(mrun[e] - mnew);
      float p0 = __expf(sc[0][e] - mnew);
      float p1 = __expf(sc[1][e] - mnew);
      sc[0][e] = p0; sc[1][e] = p1;
      float ps = p0 + p1;
#pragma unroll
      for (int off = 1; off <= 8; off <<= 1)
        ps += __shfl_xor(ps, off, 32);
      lrun[e] = lrun[e] * scale + ps;
      mrun[e] = mnew;
#pragma unroll
      for (int dt = 0; dt < 4; ++dt) oacc[dt][e] *= scale;
    }

    // ---- stage P (C->A layout, only remaining conversion) + V chunk (pure copy) ----
#pragma unroll
    for (int hh = 0; hh < 2; ++hh)
#pragma unroll
      for (int e = 0; e < 8; ++e)
        sP[(halo * 8 + e) * 40 + hh * 16 + lcol] = f2bf(sc[hh][e]);
#pragma unroll
    for (int i = 0; i < 8; ++i) {
      int idx = lane + i * 32;           // 256 x v8bf of the 32x64 bf16 V chunk
      int sl = idx >> 3;                 // key row in chunk
      int d8 = idx & 7;                  // 8-elem group in head dim
      v8bf v = *(const v8bf*)&V[((size_t)(b * TT + s0 + sl)) * DH + d8 * 8];
#pragma unroll
      for (int q = 0; q < 8; ++q) sV[(d8 * 8 + q) * 40 + sl] = v[q];
    }
    __syncthreads();

    v16bf ap = ld_frag(&sP[lcol * 40 + halo * 8], &sP[lcol * 40 + halo * 8 + 16]);
#pragma unroll
    for (int dt = 0; dt < 4; ++dt) {
      int d = dt * 16 + lcol;
      v16bf bv = ld_frag(&sV[d * 40 + halo * 8], &sV[d * 40 + halo * 8 + 16]);
      oacc[dt] = wmma_bf16(ap, bv, oacc[dt]);
    }
    __syncthreads();
  }

  // ---- epilogue: normalize, store bf16 [B,T,H*Dh] ----
#pragma unroll
  for (int dt = 0; dt < 4; ++dt)
#pragma unroll
    for (int e = 0; e < 8; ++e) {
      int row = qt * 16 + halo * 8 + e;
      int col = h * DH + dt * 16 + lcol;
      O[((size_t)(b * TT + row)) * DM + col] = f2bf(oacc[dt][e] / lrun[e]);
    }
}

// ---------------------------------------------------------------------------
extern "C" void kernel_launch(void* const* d_in, const int* in_sizes, int n_in,
                              void* d_out, int out_size, void* d_ws, size_t ws_size,
                              hipStream_t stream) {
  const float* x  = (const float*)d_in[0];
  const float* Wq = (const float*)d_in[1];
  const float* bq = (const float*)d_in[2];
  const float* Wk = (const float*)d_in[3];
  const float* bk = (const float*)d_in[4];
  const float* Wv = (const float*)d_in[5];
  const float* bv = (const float*)d_in[6];
  const float* Wo = (const float*)d_in[7];
  const float* bo = (const float*)d_in[8];
  float* out = (float*)d_out;

  // bf16 workspace: xb | Wqt | Wkt | Wvt | Wot | Qb | Kb | Vb | Ab
  __bf16* xb  = (__bf16*)d_ws;
  __bf16* Wqt = xb  + (size_t)MM * DM;
  __bf16* Wkt = Wqt + (size_t)DM * DM;
  __bf16* Wvt = Wkt + (size_t)DM * DH;
  __bf16* Wot = Wvt + (size_t)DM * DH;
  __bf16* Qb  = Wot + (size_t)DM * DM;
  __bf16* Kb  = Qb  + (size_t)MM * DM;
  __bf16* Vb  = Kb  + (size_t)MM * DH;
  __bf16* Ab  = Vb  + (size_t)MM * DH;

  dim3 blk(256);
  // one-time conversions
  cvt_f32_bf16<<<(MM * DM) / 4096, blk, 0, stream>>>(x, xb);
  cvt_transpose_bf16<<<dim3(DM / 32, DM / 32), blk, 0, stream>>>(Wq, Wqt, DM, DM);
  cvt_transpose_bf16<<<dim3(DH / 32, DM / 32), blk, 0, stream>>>(Wk, Wkt, DM, DH);
  cvt_transpose_bf16<<<dim3(DH / 32, DM / 32), blk, 0, stream>>>(Wv, Wvt, DM, DH);
  cvt_transpose_bf16<<<dim3(DM / 32, DM / 32), blk, 0, stream>>>(Wo, Wot, DM, DM);
  // projections (Q pre-scaled by 1/sqrt(Dh)), attention, output projection
  gemm_bias_wmma<128, true><<<dim3(DM / 128, MM / 128), blk, 0, stream>>>(
      xb, Wqt, bq, Qb, MM, DM, DM, 0.125f);
  gemm_bias_wmma<64, true><<<dim3(1, MM / 128), blk, 0, stream>>>(
      xb, Wkt, bk, Kb, MM, DH, DM, 1.0f);
  gemm_bias_wmma<64, true><<<dim3(1, MM / 128), blk, 0, stream>>>(
      xb, Wvt, bv, Vb, MM, DH, DM, 1.0f);
  mqa_flash_attn<<<dim3(TT / 16, NH, BB), dim3(32), 0, stream>>>(Qb, Kb, Vb, Ab);
  gemm_bias_wmma<128, false><<<dim3(DM / 128, MM / 128), blk, 0, stream>>>(
      Ab, Wot, bo, out, MM, DM, DM, 1.0f);
}
